// StickBreakingAttention_29515015258370
// MI455X (gfx1250) — matrix operand order repaired and checked by reading, not compiled, not run on hardware
//
#include <hip/hip_runtime.h>

typedef __bf16 bf16;
typedef bf16  v4bf  __attribute__((ext_vector_type(4)));
typedef bf16  v8bf  __attribute__((ext_vector_type(8)));
typedef bf16  v16bf __attribute__((ext_vector_type(16)));
typedef float v8f   __attribute__((ext_vector_type(8)));
typedef int   i32x4 __attribute__((ext_vector_type(4)));

#define N_SEQ 2048
#define EMB   1024
#define HEADS 16
#define DHEAD 64

// ---------------- bf16 converts (RNE) ------------------------------------------------------
__device__ __forceinline__ bf16 f2bf(float f) {
    unsigned u = __float_as_uint(f);
    unsigned r = (u + 0x7FFFu + ((u >> 16) & 1u)) >> 16;
    unsigned short s = (unsigned short)r;
    bf16 o; __builtin_memcpy(&o, &s, 2); return o;
}
__device__ __forceinline__ float bf2f(bf16 b) {
    unsigned short s; __builtin_memcpy(&s, &b, 2);
    return __uint_as_float(((unsigned)s) << 16);
}

// ---------------- async global->LDS copy (CDNA5), guarded with sync fallback ----------------
#if defined(__has_builtin)
#  if __has_builtin(__builtin_amdgcn_global_load_async_to_lds_b128)
#    define HAVE_ASYNC_LDS 1
#  endif
#endif
#ifndef HAVE_ASYNC_LDS
#  define HAVE_ASYNC_LDS 0
#endif

#if HAVE_ASYNC_LDS
typedef int gvec4i __attribute__((vector_size(16)));   // matches builtin's pointee type exactly
#define ASG __attribute__((address_space(1)))
#define ASL __attribute__((address_space(3)))
__device__ __forceinline__ void cp16_to_lds(bf16* dst, const bf16* src) {
    __builtin_amdgcn_global_load_async_to_lds_b128((ASG gvec4i*)src, (ASL gvec4i*)dst, 0, 0);
}
__device__ __forceinline__ void cp_wait() {
#if __has_builtin(__builtin_amdgcn_s_wait_asynccnt)
    __builtin_amdgcn_s_wait_asynccnt(0);
#else
    asm volatile("s_wait_asynccnt 0x0" ::: "memory");
#endif
}
#else
__device__ __forceinline__ void cp16_to_lds(bf16* dst, const bf16* src) {
    *(i32x4*)dst = *(const i32x4*)src;
}
__device__ __forceinline__ void cp_wait() {}
#endif

// ---------------- fp32 -> bf16 elementwise (count % 4 == 0) --------------------------------
__global__ void f2bf_kernel(const float* __restrict__ in, bf16* __restrict__ out, long count) {
    long idx = ((long)blockIdx.x * blockDim.x + threadIdx.x) * 4;
    long stride = (long)gridDim.x * blockDim.x * 4;
    for (long i = idx; i < count; i += stride) {
        float4 f = *(const float4*)(in + i);
        v4bf o;
        o.x = f2bf(f.x); o.y = f2bf(f.y); o.z = f2bf(f.z); o.w = f2bf(f.w);
        *(v4bf*)(out + i) = o;
    }
}

// ---------------- fp32 (rows x cols) -> bf16 transpose (cols x rows) -----------------------
__global__ void transpose_f2bf_kernel(const float* __restrict__ in, bf16* __restrict__ out,
                                      int rows, int cols) {
    __shared__ float tile[32][33];
    int bx = blockIdx.x * 32, by = blockIdx.y * 32;
    int tx = threadIdx.x, ty = threadIdx.y;          // block (32, 8)
#pragma unroll
    for (int k = 0; k < 32; k += 8)
        tile[ty + k][tx] = in[(long)(by + ty + k) * cols + (bx + tx)];
    __syncthreads();
#pragma unroll
    for (int k = 0; k < 32; k += 8)
        out[(long)(bx + ty + k) * rows + (by + tx)] = f2bf(tile[tx][ty + k]);
}

// ---------------- batched WMMA GEMM: C = alpha * (A @ B^T) ---------------------------------
// A: bf16 [M x K] (lda), B: bf16 [N x K] (ldb), C: fp32 or bf16 (ldc).
// 256 threads = 8 waves. Block tile 64x128, BK=64, double-buffered LDS with async staging.
// Wave tile 32x32 (2x2 WMMA f32 accumulators), 8 v_wmma per K-slab per wave.
template <bool STORE_BF16>
__global__ __launch_bounds__(256)
void gemm_wmma(const bf16* __restrict__ A, long lda, long strideA,
               const bf16* __restrict__ B, long ldb, long strideB,
               void* __restrict__ C, long ldc, long strideC,
               int M, int N, int K, float alpha) {
    __shared__ __attribute__((aligned(16))) bf16 sA[2][64][72];   // 144B rows, 16B-aligned chunks
    __shared__ __attribute__((aligned(16))) bf16 sB[2][128][72];

    const int tid  = threadIdx.x;
    const int lane = tid & 31;
    const int wave = tid >> 5;
    const int wr   = wave >> 2;        // 0..1
    const int wc   = wave & 3;         // 0..3
    const int l16  = lane & 15;
    const int lh   = lane >> 4;        // lane half

    const long blockM = (long)blockIdx.y * 64;
    const long blockN = (long)blockIdx.x * 128;
    const bf16* Ab = A + (long)blockIdx.z * strideA;
    const bf16* Bb = B + (long)blockIdx.z * strideB;
    const long cOff = (long)blockIdx.z * strideC;

    auto stage = [&](int buf, int kk) {
#pragma unroll
        for (int it = 0; it < 2; ++it) {          // A tile: 64 x 64 bf16 = 512 x 16B chunks
            int idx = tid + it * 256;
            int r = idx >> 3, c = idx & 7;
            long gr = blockM + r; if (gr > M - 1) gr = M - 1;
            cp16_to_lds(&sA[buf][r][c * 8], Ab + gr * lda + kk + c * 8);
        }
#pragma unroll
        for (int it = 0; it < 4; ++it) {          // B tile: 128 x 64 bf16 = 1024 x 16B chunks
            int idx = tid + it * 256;
            int r = idx >> 3, c = idx & 7;
            long gr = blockN + r; if (gr > N - 1) gr = N - 1;
            cp16_to_lds(&sB[buf][r][c * 8], Bb + gr * ldb + kk + c * 8);
        }
    };

    v8f acc[2][2] = {};
    stage(0, 0);
    int cur = 0;
    for (int kk = 0; kk < K; kk += 64) {
        cp_wait();                 // own async copies into `cur` landed
        __syncthreads();           // everyone's copies landed; prior reads of cur^1 finished
        if (kk + 64 < K) stage(cur ^ 1, kk + 64);
#pragma unroll
        for (int ks = 0; ks < 64; ks += 32) {
            v16bf afr[2], bfr[2];
#pragma unroll
            for (int mt = 0; mt < 2; ++mt) {      // A frag: 16-bit 16x32 layout
                const bf16* bse = &sA[cur][wr * 32 + mt * 16 + l16][ks + lh * 16];
                v8bf lo = *(const v8bf*)(bse);
                v8bf hi = *(const v8bf*)(bse + 8);
                afr[mt] = __builtin_shufflevector(lo, hi, 0,1,2,3,4,5,6,7,8,9,10,11,12,13,14,15);
            }
#pragma unroll
            for (int nt = 0; nt < 2; ++nt) {      // B frag: 16-bit 32x16 layout
                const bf16* bse = &sB[cur][wc * 32 + nt * 16 + l16][ks + lh * 16];
                v8bf lo = *(const v8bf*)(bse);
                v8bf hi = *(const v8bf*)(bse + 8);
                bfr[nt] = __builtin_shufflevector(lo, hi, 0,1,2,3,4,5,6,7,8,9,10,11,12,13,14,15);
            }
#pragma unroll
            for (int mt = 0; mt < 2; ++mt)
#pragma unroll
                for (int nt = 0; nt < 2; ++nt)
                    acc[mt][nt] = __builtin_amdgcn_wmma_f32_16x16x32_bf16(
                        false, afr[mt], false, bfr[nt], (short)0, acc[mt][nt], false, false);
        }
        cur ^= 1;
    }

    // epilogue: 16x16 f32 D layout — lane half selects M+8, VGPR r = M row
#pragma unroll
    for (int mt = 0; mt < 2; ++mt) {
#pragma unroll
        for (int nt = 0; nt < 2; ++nt) {
            long col = blockN + wc * 32 + nt * 16 + l16;
            if (col >= N) continue;
            long row0 = blockM + wr * 32 + mt * 16 + lh * 8;
#pragma unroll
            for (int r = 0; r < 8; ++r) {
                long row = row0 + r;
                if (row < M) {
                    float val = acc[mt][nt][r] * alpha;
                    if (STORE_BF16)
                        ((bf16*)C)[cOff + row * ldc + col] = f2bf(val);
                    else
                        ((float*)C)[cOff + row * ldc + col] = val;
                }
            }
        }
    }
}

// ---------------- stick-breaking row scan (wave32 shuffle scan) -----------------------------
// One block (256 thr = 8 waves) per row (h,i). logits bf16 (h,n,n) pre-scaled.
__global__ __launch_bounds__(256)
void sb_scan_kernel(const bf16* __restrict__ logits, bf16* __restrict__ att,
                    float* __restrict__ rem, int n) {
    const int row = blockIdx.x;               // h*n + i
    const int i   = row & (n - 1);            // n power of two
    const long base = (long)row * n;
    const int tid  = threadIdx.x;
    const int lane = tid & 31;
    const int w    = tid >> 5;
    __shared__ float swave[8];

    float lb[N_SEQ / 256], lz[N_SEQ / 256];
    float tsum = 0.f;
#pragma unroll
    for (int c = 0; c < N_SEQ / 256; ++c) {
        int j = c * 256 + tid;
        float s = bf2f(logits[base + j]);
        float t = log1pf(__expf(-fabsf(s)));
        bool masked = (j >= i);
        lz[c] = masked ? -100000.0f : (fminf(s, 0.f) - t);    // log_sigmoid(s)
        lb[c] = masked ? 0.0f       : (fminf(-s, 0.f) - t);   // log_sigmoid(-s)
        tsum += lb[c];
    }
    // block reduce -> total
    float v = tsum;
#pragma unroll
    for (int off = 16; off > 0; off >>= 1) v += __shfl_xor(v, off, 32);
    if (lane == 0) swave[w] = v;
    __syncthreads();
    float total = 0.f;
#pragma unroll
    for (int ww = 0; ww < 8; ++ww) total += swave[ww];

    float running = 0.f, asum = 0.f;
#pragma unroll
    for (int c = 0; c < N_SEQ / 256; ++c) {
        // wave inclusive scan
        float incl = lb[c];
#pragma unroll
        for (int off = 1; off < 32; off <<= 1) {
            float y = __shfl_up(incl, off, 32);
            if (lane >= off) incl += y;
        }
        float wtot = __shfl(incl, 31, 32);
        __syncthreads();                     // prior swave reads done
        if (lane == 0) swave[w] = wtot;
        __syncthreads();
        float wpre = 0.f, ctot = 0.f;
#pragma unroll
        for (int ww = 0; ww < 8; ++ww) {
            float t2 = swave[ww];
            ctot += t2;
            if (ww < w) wpre += t2;
        }
        int j = c * 256 + tid;
        float av = __expf(lz[c] + total - (running + wpre + incl));
        att[base + j] = f2bf(av);
        asum += av;
        running += ctot;
    }
    // rem = 1 - sum(att)
    float r2 = asum;
#pragma unroll
    for (int off = 16; off > 0; off >>= 1) r2 += __shfl_xor(r2, off, 32);
    __syncthreads();
    if (lane == 0) swave[w] = r2;
    __syncthreads();
    if (tid == 0) {
        float t = 0.f;
#pragma unroll
        for (int ww = 0; ww < 8; ++ww) t += swave[ww];
        rem[row] = 1.0f - t;
    }
}

// ---------------- o = of + rem*v, convert to bf16 -------------------------------------------
__global__ void fuse_o_kernel(const float* __restrict__ of, const float* __restrict__ v,
                              const float* __restrict__ rem, bf16* __restrict__ ob,
                              int n, int e, int d) {
    long total = (long)n * e;
    long idx = (long)blockIdx.x * blockDim.x + threadIdx.x;
    long stride = (long)gridDim.x * blockDim.x;
    for (long k = idx; k < total; k += stride) {
        long r = k / e; int c = (int)(k - r * e); int h = c / d;
        ob[k] = f2bf(of[k] + rem[(long)h * n + r] * v[k]);
    }
}

// ---------------- host side -----------------------------------------------------------------
static void launch_gemm(const bf16* A, long lda, long sA, const bf16* B, long ldb, long sB,
                        void* C, long ldc, long sC, int M, int N, int K, int batch,
                        float alpha, bool bf16out, hipStream_t s) {
    dim3 grid((N + 127) / 128, (M + 63) / 64, batch), block(256);
    if (bf16out)
        gemm_wmma<true><<<grid, block, 0, s>>>(A, lda, sA, B, ldb, sB, C, ldc, sC, M, N, K, alpha);
    else
        gemm_wmma<false><<<grid, block, 0, s>>>(A, lda, sA, B, ldb, sB, C, ldc, sC, M, N, K, alpha);
}

extern "C" void kernel_launch(void* const* d_in, const int* in_sizes, int n_in,
                              void* d_out, int out_size, void* d_ws, size_t ws_size,
                              hipStream_t stream) {
    const int n = N_SEQ, e = EMB, h = HEADS, d = DHEAD;
    const float* x  = (const float*)d_in[0];
    const float* Wq = (const float*)d_in[1];
    const float* Wk = (const float*)d_in[2];
    const float* Wv = (const float*)d_in[3];
    const float* Wo = (const float*)d_in[4];
    float* out = (float*)d_out;

    char* ws = (char*)d_ws;
    size_t off = 0;
    auto carve = [&](size_t bytes) { char* p = ws + off; off += (bytes + 255) & ~(size_t)255; return p; };
    bf16*  xb   = (bf16*)carve((size_t)n * e * 2);
    bf16*  wqb  = (bf16*)carve((size_t)e * e * 2);
    bf16*  wkb  = (bf16*)carve((size_t)e * e * 2);
    bf16*  wvb  = (bf16*)carve((size_t)e * e * 2);
    bf16*  wob  = (bf16*)carve((size_t)e * e * 2);
    float* qf   = (float*)carve((size_t)n * e * 4);
    float* kf   = (float*)carve((size_t)n * e * 4);
    float* vf   = (float*)carve((size_t)n * e * 4);
    bf16*  qb   = (bf16*)carve((size_t)n * e * 2);
    bf16*  kb   = (bf16*)carve((size_t)n * e * 2);
    bf16*  vtb  = (bf16*)carve((size_t)e * n * 2);
    float* of   = (float*)carve((size_t)n * e * 4);
    bf16*  ob   = (bf16*)carve((size_t)n * e * 2);
    float* rem  = (float*)carve((size_t)h * n * 4);
    bf16*  lg   = (bf16*)carve((size_t)h * n * n * 2);
    bf16*  att  = (bf16*)carve((size_t)h * n * n * 2);
    (void)ws_size; (void)n_in; (void)in_sizes; (void)out_size;

    dim3 cvB(256), cvG(1024);
    f2bf_kernel<<<cvG, cvB, 0, stream>>>(x,  xb,  (long)n * e);
    f2bf_kernel<<<cvG, cvB, 0, stream>>>(Wq, wqb, (long)e * e);
    f2bf_kernel<<<cvG, cvB, 0, stream>>>(Wk, wkb, (long)e * e);
    f2bf_kernel<<<cvG, cvB, 0, stream>>>(Wv, wvb, (long)e * e);
    f2bf_kernel<<<cvG, cvB, 0, stream>>>(Wo, wob, (long)e * e);

    // projections: q/k/v = x @ W^T (fp32 out)
    launch_gemm(xb, e, 0, wqb, e, 0, qf, e, 0, n, e, e, 1, 1.0f, false, stream);
    launch_gemm(xb, e, 0, wkb, e, 0, kf, e, 0, n, e, e, 1, 1.0f, false, stream);
    launch_gemm(xb, e, 0, wvb, e, 0, vf, e, 0, n, e, e, 1, 1.0f, false, stream);

    f2bf_kernel<<<cvG, cvB, 0, stream>>>(qf, qb, (long)n * e);
    f2bf_kernel<<<cvG, cvB, 0, stream>>>(kf, kb, (long)n * e);
    transpose_f2bf_kernel<<<dim3(e / 32, n / 32), dim3(32, 8), 0, stream>>>(vf, vtb, n, e);

    // per-head logits = scale * q_h @ k_h^T -> bf16 (h, n, n)
    launch_gemm(qb, e, d, kb, e, d, lg, n, (long)n * n, n, n, d, h, 0.125f, true, stream);

    // stick-breaking scan -> att bf16, rem fp32
    sb_scan_kernel<<<h * n, 256, 0, stream>>>(lg, att, rem, n);

    // o_h = att_h @ v_h  (att [n x n] @ vT_h [d x n]^T) -> fp32 (n, e) layout
    launch_gemm(att, n, (long)n * n, vtb, n, (long)d * n, of, e, d, n, d, n, h, 1.0f, false, stream);

    // o += rem * v ; convert to bf16
    fuse_o_kernel<<<cvG, cvB, 0, stream>>>(of, vf, rem, ob, n, e, d);

    // final projection: out = o @ Wo^T (fp32)
    launch_gemm(ob, e, 0, wob, e, 0, out, e, 0, n, e, e, 1, 1.0f, false, stream);
}
